// LapSWDLoss_1357209666248
// MI455X (gfx1250) — compile-verified, matrix operand order checked by reading, not run on hardware
//
#include <hip/hip_runtime.h>
#include <math.h>

#define BATCH 128
#define CH    3
#define NPROJ 512
#define KDIM  147
#define KPAD  160
#define MPAD  16384

typedef __attribute__((ext_vector_type(16))) __bf16 v16bf;
typedef __attribute__((ext_vector_type(8)))  __bf16 v8bf;
typedef __attribute__((ext_vector_type(8)))  float  v8f;
typedef int v4i __attribute__((__vector_size__(16)));   // matches builtin param

// gfx1250 async global->LDS path (ASYNCcnt-tracked), guarded per-toolchain
#if defined(__has_builtin)
#if __has_builtin(__builtin_amdgcn_global_load_async_to_lds_b128)
#define HAVE_ASYNC_LDS 1
#endif
#if __has_builtin(__builtin_amdgcn_s_wait_asynccnt)
#define HAVE_WAIT_ASYNC 1
#endif
#endif

__device__ __forceinline__ void wait_asynccnt0() {
#if defined(HAVE_WAIT_ASYNC)
  __builtin_amdgcn_s_wait_asynccnt(0);
#else
  asm volatile("s_wait_asynccnt 0x0" ::: "memory");
#endif
}

__constant__ float c_kw[5] = {1.f/16.f, 4.f/16.f, 6.f/16.f, 4.f/16.f, 1.f/16.f};

__device__ __forceinline__ unsigned pcg(unsigned x) {
  x = x * 747796405u + 2891336453u;
  unsigned w = ((x >> ((x >> 28) + 4u)) ^ x) * 277803737u;
  return (w >> 22) ^ w;
}
__device__ __forceinline__ float u01(unsigned h) {
  return ((float)h + 0.5f) * (1.0f / 4294967296.0f);
}
__device__ __forceinline__ unsigned short f2bf(float f) {
  unsigned u = __float_as_uint(f);
  u += 0x7FFFu + ((u >> 16) & 1u);          // round-to-nearest-even
  return (unsigned short)(u >> 16);
}

// ---------------------------------------------------------------- utilities
__global__ void k_zero(float* p, int n) {
  int i = blockIdx.x * blockDim.x + threadIdx.x;
  if (i < n) p[i] = 0.f;
}

// ------------------------------------------------- 5x5 gaussian, stride 2
__global__ void k_gauss_down(const float* __restrict__ in, float* __restrict__ out,
                             int W, int Wo, int total) {
  int idx = blockIdx.x * blockDim.x + threadIdx.x;
  if (idx >= total) return;
  int q  = idx % Wo;
  int p  = (idx / Wo) % Wo;
  int bc = idx / (Wo * Wo);
  const float* src = in + (size_t)bc * W * W;
  float acc = 0.f;
#pragma unroll
  for (int i = 0; i < 5; ++i) {
    int r = 2 * p + i - 2;
    if (r < 0 || r >= W) continue;
    float wy = c_kw[i];
#pragma unroll
    for (int j = 0; j < 5; ++j) {
      int c = 2 * q + j - 2;
      if (c < 0 || c >= W) continue;
      acc += wy * c_kw[j] * src[r * W + c];
    }
  }
  out[idx] = acc;
}

// ---------------------------------------------- random patch positions
__global__ void k_positions(int* rows, int* cols, int n_desc, int navail,
                            int Wm6, unsigned seed) {
  int d = threadIdx.x;
  if (d >= n_desc) return;
  unsigned h = pcg(seed * 9781u + (unsigned)d * 2654435761u);
  int idx = (int)(h % (unsigned)navail);
  rows[d] = idx / Wm6;
  cols[d] = idx % Wm6;
}

// -------- random normal projections, per-column std-normalized (ddof=1), bf16
__global__ void k_proj_rand(unsigned short* __restrict__ Bm, unsigned seed) {
  int n = blockIdx.x;             // column 0..511
  int k = threadIdx.x;            // row 0..159
  __shared__ float s1[KPAD], s2[KPAD], sstd;
  float v = 0.f;
  if (k < KDIM) {
    unsigned base = seed * 1315423911u + (unsigned)(k * NPROJ + n);
    float a = u01(pcg(base));
    float b = u01(pcg(base ^ 0x9E3779B9u));
    v = sqrtf(-2.f * logf(a)) * cosf(6.28318530718f * b);
  }
  s1[k] = v; s2[k] = v * v;
  __syncthreads();
  if (k == 0) {
    float s = 0.f, sq = 0.f;
    for (int i = 0; i < KDIM; ++i) { s += s1[i]; sq += s2[i]; }
    float var = (sq - s * s / (float)KDIM) / (float)(KDIM - 1);
    sstd = sqrtf(fmaxf(var, 1e-20f));
  }
  __syncthreads();
  Bm[(size_t)k * NPROJ + n] = (k < KDIM) ? f2bf(v / sstd) : (unsigned short)0;
}

// ------------- gather 7x7 patches; laplacian computed on the fly; raw f32 out
__global__ void k_extract(const float* __restrict__ gcur,
                          const float* __restrict__ gnext,  // null for last layer
                          const int* __restrict__ rows, const int* __restrict__ cols,
                          float* __restrict__ pout, float* __restrict__ stats,
                          int W, int n_desc) {
  int bd = blockIdx.x;
  int d  = bd % n_desc;
  int b  = bd / n_desc;
  int t  = threadIdx.x;           // 0..191, active < 147
  __shared__ float ls[6];
  if (t < 6) ls[t] = 0.f;
  __syncthreads();
  float val = 0.f;
  int c = 0;
  if (t < KDIM) {
    c = t / 49;
    int pos = t % 49, pr = pos / 7, pc = pos % 7;
    int r = rows[d] + pr, q = cols[d] + pc;
    const float* src = gcur + ((size_t)(b * CH + c)) * W * W;
    val = src[r * W + q];
    if (gnext) {
      int Wn = W >> 1;
      const float* nx = gnext + ((size_t)(b * CH + c)) * Wn * Wn;
      float acc = 0.f;
#pragma unroll
      for (int i = 0; i < 5; ++i) {
        int rr = r + i - 2;
        if (rr < 0 || rr >= W) continue;
        float wy = c_kw[i];
#pragma unroll
        for (int j = 0; j < 5; ++j) {
          int qq = q + j - 2;
          if (qq < 0 || qq >= W) continue;
          acc += wy * c_kw[j] * nx[(rr >> 1) * Wn + (qq >> 1)];
        }
      }
      val -= acc;                 // laplacian = g_i - smooth(up(g_{i+1}))
    }
    pout[(size_t)bd * KDIM + t] = val;
    atomicAdd(&ls[c], val);
    atomicAdd(&ls[3 + c], val * val);
  }
  __syncthreads();
  if (t < 6) atomicAdd(&stats[t], ls[t]);
}

// -------------- normalize by per-channel mean/std(ddof=1), pack bf16, pad K
__global__ void k_norm(const float* __restrict__ pin, const float* __restrict__ stats,
                       unsigned short* __restrict__ Aout, int Mreal) {
  int idx = blockIdx.x * blockDim.x + threadIdx.x;
  int total = Mreal * KPAD;
  if (idx >= total) return;
  int k = idx % KPAD;
  int m = idx / KPAD;
  unsigned short o = 0;
  if (k < KDIM) {
    int c = k / 49;
    float cnt  = (float)Mreal * 49.f;
    float mean = stats[c] / cnt;
    float var  = (stats[3 + c] - cnt * mean * mean) / (cnt - 1.f);
    float sd   = sqrtf(fmaxf(var, 0.f));
    o = f2bf((pin[(size_t)m * KDIM + k] - mean) / (sd + 1e-8f));
  }
  Aout[(size_t)m * KPAD + k] = o;
}

// ----- WMMA bf16 GEMM: [M,160] x [160,512] -> col-major C
// Block = 8 waves = 128M x 64N tile. Each wave: 1 A-fragment reused across
// 4 N-subtiles (4 WMMAs per K-step, 20 per wave) -> 4x less A traffic.
__global__ void k_gemm(const unsigned short* __restrict__ A,
                       const unsigned short* __restrict__ Bm,
                       float* __restrict__ Cout, int Mreal, int Mpad) {
  const int lane = threadIdx.x & 31;
  const int wave = threadIdx.x >> 5;
  int n0 = (blockIdx.x & 7) * 64;             // 512/64 = 8 N groups
  int mg = blockIdx.x >> 3;
  int m0 = (mg * 8 + wave) * 16;

  __shared__ __align__(16) unsigned short ldsB[64 * KPAD];  // B tile, [n][k]
  for (int i = threadIdx.x; i < 64 * KPAD; i += blockDim.x) {
    int n = i / KPAD, k = i % KPAD;
    ldsB[n * KPAD + k] = Bm[(size_t)k * NPROJ + n0 + n];
  }
  __syncthreads();

  int half = lane >> 4;
  int lm   = lane & 15;
  const unsigned short* arow = A + (size_t)(m0 + lm) * KPAD;
  __builtin_prefetch(arow, 0, 0);

  v8f acc[4];
#pragma unroll
  for (int t = 0; t < 4; ++t) acc[t] = (v8f){0.f,0.f,0.f,0.f,0.f,0.f,0.f,0.f};

#pragma unroll
  for (int kb = 0; kb < KPAD; kb += 32) {
    // 16-bit A layout: lane(0-15)=rows M, half selects K sub-blocks; 16B chunks
    v8bf alo = *(const v8bf*)(arow + kb + half * 8);
    v8bf ahi = *(const v8bf*)(arow + kb + 16 + half * 8);
    v16bf av;
#pragma unroll
    for (int e = 0; e < 8; ++e) { av[e] = alo[e]; av[e + 8] = ahi[e]; }
#pragma unroll
    for (int t = 0; t < 4; ++t) {
      const unsigned short* brow = ldsB + (t * 16 + lm) * KPAD;
      v8bf blo = *(const v8bf*)(brow + kb + half * 8);
      v8bf bhi = *(const v8bf*)(brow + kb + 16 + half * 8);
      v16bf bv;
#pragma unroll
      for (int e = 0; e < 8; ++e) { bv[e] = blo[e]; bv[e + 8] = bhi[e]; }
      acc[t] = __builtin_amdgcn_wmma_f32_16x16x32_bf16(false, av, false, bv,
                                                       (short)0, acc[t], false, false);
    }
  }
  // C 16x16 f32: VGPR r -> M = r + 8*half ; N = lane%16. Column-major out.
#pragma unroll
  for (int t = 0; t < 4; ++t) {
    float* col = Cout + (size_t)(n0 + t * 16 + lm) * Mpad + m0 + half * 8;
#pragma unroll
    for (int r = 0; r < 8; ++r) col[r] = acc[t][r];
  }
}

// ----------------------- per-column in-LDS bitonic sort of 16384 elements
__global__ void k_sort(float* __restrict__ p1, float* __restrict__ p2,
                       int Mreal, int Mpad) {
  extern __shared__ float s[];
  int col = blockIdx.x & (NPROJ - 1);
  float* base = ((blockIdx.x < NPROJ) ? p1 : p2) + (size_t)col * Mpad;
#if defined(HAVE_ASYNC_LDS)
  // stream column into LDS without a VGPR round-trip (ASYNCcnt path)
  for (int i = threadIdx.x * 4; i < Mpad; i += blockDim.x * 4) {
    __builtin_amdgcn_global_load_async_to_lds_b128(
        (__attribute__((address_space(1))) v4i*)(base + i),
        (__attribute__((address_space(3))) v4i*)(&s[i]), 0, 0);
  }
  wait_asynccnt0();
#else
  for (int i = threadIdx.x; i < Mpad; i += blockDim.x) s[i] = base[i];
#endif
  __syncthreads();
  for (int i = threadIdx.x; i < Mpad; i += blockDim.x)
    if (i >= Mreal) s[i] = 3.402823466e38f;
  __syncthreads();
  for (int k = 2; k <= Mpad; k <<= 1) {
    for (int j = k >> 1; j > 0; j >>= 1) {
      for (int i = threadIdx.x; i < Mpad; i += blockDim.x) {
        int ixj = i ^ j;
        if (ixj > i) {
          bool up = ((i & k) == 0);
          float a = s[i], b = s[ixj];
          if ((a > b) == up) { s[i] = b; s[ixj] = a; }
        }
      }
      __syncthreads();
    }
  }
  for (int i = threadIdx.x; i < Mpad; i += blockDim.x) base[i] = s[i];
}

// --------------------------- sum |sorted1 - sorted2| into layer accumulator
__global__ void k_absdiff(const float* __restrict__ p1, const float* __restrict__ p2,
                          int total, float* __restrict__ acc) {
  __shared__ float sh[256];
  float s = 0.f;
  for (int i = blockIdx.x * blockDim.x + threadIdx.x; i < total;
       i += gridDim.x * blockDim.x)
    s += fabsf(p1[i] - p2[i]);
  sh[threadIdx.x] = s;
  __syncthreads();
  for (int o = 128; o > 0; o >>= 1) {
    if (threadIdx.x < o) sh[threadIdx.x] += sh[threadIdx.x + o];
    __syncthreads();
  }
  if (threadIdx.x == 0) atomicAdd(acc, sh[0]);
}

__global__ void k_final(float* out, const float* acc, float i0, float i1,
                        float i2, float i3, float i4) {
  if (threadIdx.x == 0 && blockIdx.x == 0) {
    float m = acc[0] * i0 + acc[1] * i1 + acc[2] * i2 + acc[3] * i3 + acc[4] * i4;
    out[0] = m * (1000.f / 5.f);
  }
}

// ============================================================== host driver
extern "C" void kernel_launch(void* const* d_in, const int* in_sizes, int n_in,
                              void* d_out, int out_size, void* d_ws, size_t ws_size,
                              hipStream_t stream) {
  const float* x = (const float*)d_in[0];
  const float* y = (const float*)d_in[1];
  float* out = (float*)d_out;

  size_t off = 0;
  auto alloc = [&](size_t bytes) -> char* {
    char* r = (char*)d_ws + off;
    off += (bytes + 255) & ~(size_t)255;
    return r;
  };

  int dim[5] = {256, 128, 64, 32, 16};
  const float* gX[5]; const float* gY[5];
  gX[0] = x; gY[0] = y;
  float* gXb[5] = {nullptr}; float* gYb[5] = {nullptr};
  for (int l = 1; l < 5; ++l) {
    size_t bytes = (size_t)BATCH * CH * dim[l] * dim[l] * sizeof(float);
    gXb[l] = (float*)alloc(bytes); gX[l] = gXb[l];
    gYb[l] = (float*)alloc(bytes); gY[l] = gYb[l];
  }
  float* P1   = (float*)alloc((size_t)MPAD * KDIM * sizeof(float));
  float* P2   = (float*)alloc((size_t)MPAD * KDIM * sizeof(float));
  unsigned short* A1 = (unsigned short*)alloc((size_t)MPAD * KPAD * 2);
  unsigned short* A2 = (unsigned short*)alloc((size_t)MPAD * KPAD * 2);
  unsigned short* Bm = (unsigned short*)alloc((size_t)KPAD * NPROJ * 2);
  float* PR1  = (float*)alloc((size_t)MPAD * NPROJ * sizeof(float));
  float* PR2  = (float*)alloc((size_t)MPAD * NPROJ * sizeof(float));
  int*   rows = (int*)alloc(128 * sizeof(int));
  int*   cols = (int*)alloc(128 * sizeof(int));
  float* stats = (float*)alloc(32 * sizeof(float));   // [0..5]=x, [8..13]=y
  float* acc   = (float*)alloc(8 * sizeof(float));

  k_zero<<<1, 32, 0, stream>>>(acc, 8);

  // gaussian pyramids
  for (int l = 1; l < 5; ++l) {
    int total = BATCH * CH * dim[l] * dim[l];
    int g = (total + 255) / 256;
    k_gauss_down<<<g, 256, 0, stream>>>(gX[l - 1], gXb[l], dim[l - 1], dim[l], total);
    k_gauss_down<<<g, 256, 0, stream>>>(gY[l - 1], gYb[l], dim[l - 1], dim[l], total);
  }

  for (int l = 0; l < 5; ++l) {
    int W = dim[l];
    int nd = (l == 4) ? 100 : 128;
    int Wm6 = W - 6;
    int Mreal = BATCH * nd;

    k_positions<<<1, 128, 0, stream>>>(rows, cols, nd, Wm6 * Wm6, Wm6, 42u + l);
    k_zero<<<1, 32, 0, stream>>>(stats, 32);
    k_proj_rand<<<NPROJ, KPAD, 0, stream>>>(Bm, 1000u + l);

    const float* gn_x = (l < 4) ? gX[l + 1] : nullptr;
    const float* gn_y = (l < 4) ? gY[l + 1] : nullptr;
    k_extract<<<BATCH * nd, 192, 0, stream>>>(gX[l], gn_x, rows, cols, P1, stats, W, nd);
    k_extract<<<BATCH * nd, 192, 0, stream>>>(gY[l], gn_y, rows, cols, P2, stats + 8, W, nd);

    int gn = (Mreal * KPAD + 255) / 256;
    k_norm<<<gn, 256, 0, stream>>>(P1, stats, A1, Mreal);
    k_norm<<<gn, 256, 0, stream>>>(P2, stats + 8, A2, Mreal);

    int gg = 8 * (Mreal / 128);   // 8 N-groups x (M/128) blocks of 8 waves
    k_gemm<<<gg, 256, 0, stream>>>(A1, Bm, PR1, Mreal, MPAD);
    k_gemm<<<gg, 256, 0, stream>>>(A2, Bm, PR2, Mreal, MPAD);

    k_sort<<<2 * NPROJ, 1024, MPAD * sizeof(float), stream>>>(PR1, PR2, Mreal, MPAD);

    k_absdiff<<<4096, 256, 0, stream>>>(PR1, PR2, NPROJ * MPAD, acc + l);
  }

  float inv[5];
  for (int l = 0; l < 5; ++l) {
    int nd = (l == 4) ? 100 : 128;
    inv[l] = 1.f / ((float)NPROJ * (float)(BATCH * nd));
  }
  k_final<<<1, 32, 0, stream>>>(out, acc, inv[0], inv[1], inv[2], inv[3], inv[4]);
}
